// LSTM_91018946937276
// MI455X (gfx1250) — matrix operand order repaired and checked by reading, not compiled, not run on hardware
//
#include <hip/hip_runtime.h>
#include <hip/hip_bf16.h>
#include <stdint.h>

// Problem constants (from reference setup_inputs)
#define B_  256
#define T_  1024
#define D_  128
#define H_  256
#define FH_ 1024   // 4*H
#define C_  10

typedef __attribute__((ext_vector_type(16))) __bf16 v16bf;
typedef __attribute__((ext_vector_type(8)))  __bf16 v8bf;
typedef __attribute__((ext_vector_type(8)))  float  v8f;

__device__ __forceinline__ float sigmoidf_(float x) {
    return 1.0f / (1.0f + __expf(-x));
}
__device__ __forceinline__ float tanhf_fast(float x) {
    // tanh(x) = 2*sigmoid(2x) - 1  (keeps everything on v_exp_f32)
    return __builtin_fmaf(2.0f, sigmoidf_(2.0f * x), -1.0f);
}

// ---- WMMA fragment loaders ---------------------------------------------
// A-matrix (16-bit, 16x32 MxK) lane layout:
//   lanes 0-15  (lhalf=0): VGPR0-3 = K[kb+0..7],  VGPR4-7 = K[kb+16..23]
//   lanes 16-31 (lhalf=1): VGPR0-3 = K[kb+8..15], VGPR4-7 = K[kb+24..31]
// => per lane: two contiguous 8-element runs at kb+lhalf*8 and kb+lhalf*8+16.

__device__ __forceinline__ v16bf load_a_f32(const float* __restrict__ rowBase,
                                            int kb, int lhalf) {
    const float* p = rowBase + kb + lhalf * 8;
    v16bf a;
#pragma unroll
    for (int j = 0; j < 8; ++j) a[j] = (__bf16)p[j];
#pragma unroll
    for (int j = 0; j < 8; ++j) a[8 + j] = (__bf16)p[16 + j];
    return a;
}

__device__ __forceinline__ v16bf load_a_bf16(const __bf16* rowBase,
                                             int kb, int lhalf) {
    const __bf16* p = rowBase + kb + lhalf * 8;
    v8bf lo = *(const v8bf*)(p);
    v8bf hi = *(const v8bf*)(p + 16);
    return __builtin_shufflevector(lo, hi, 0, 1, 2, 3, 4, 5, 6, 7,
                                           8, 9, 10, 11, 12, 13, 14, 15);
}

// B-matrix (16-bit, 32x16 KxN) lane layout: N = lane&15, K order matches A:
// two 8-element runs at kb + (lane>>4)*8 and kb + (lane>>4)*8 + 16.  Weights
// are stored transposed ([N][K], K contiguous) so this is two 16-byte loads.
__device__ __forceinline__ v16bf load_b(const __bf16* wT, int K,
                                        int n_g, int kb, int lhalf) {
    const __bf16* p = wT + (size_t)n_g * K + kb + lhalf * 8;
    v8bf lo = *(const v8bf*)(p);
    v8bf hi = *(const v8bf*)(p + 16);
    return __builtin_shufflevector(lo, hi, 0, 1, 2, 3, 4, 5, 6, 7,
                                           8, 9, 10, 11, 12, 13, 14, 15);
}

// ---- Whole-sequence LSTM: one block = 16 batch rows, all T steps --------
// Batch rows are independent in the recurrence; the GEMM couples only over
// H, which lives block-locally in LDS.  16 waves/block: wave w owns the
// 16x16 output tile at hidden columns [w*16, w*16+16) for all four gates.
__global__ __launch_bounds__(512)
void lstm_seq_kernel(const float* __restrict__ x,     // [B][T][D] f32
                     const __bf16* __restrict__ wxT,  // [4H][D] bf16
                     const __bf16* __restrict__ whT,  // [4H][H] bf16
                     const float*  __restrict__ bias, // [4H]
                     const float*  __restrict__ wph,  // [H][C] f32
                     const float*  __restrict__ bp,   // [C]
                     float* __restrict__ out)         // [B][C] f32
{
    __shared__ __bf16 hbuf[2][16 * H_];   // double-buffered h tile (16 KB)

    const int tid   = threadIdx.x;
    const int lane  = tid & 31;
    const int w     = tid >> 5;           // wave 0..15
    const int lidx  = lane & 15;
    const int lhalf = lane >> 4;
    const int m0    = blockIdx.x * 16;    // batch-block origin
    const int n0    = w * 16;             // hidden-tile origin (within H)
    const int mloc  = lidx;               // A row within the block tile
    const int m     = m0 + mloc;          // global batch row for this lane

    // h_0 = 0
    for (int i = tid; i < 16 * H_; i += 512) hbuf[0][i] = (__bf16)0.0f;
    __syncthreads();

    const int n = n0 + lidx;
    const float bgv = bias[0 * H_ + n];
    const float biv = bias[1 * H_ + n];
    const float bfv = bias[2 * H_ + n];
    const float bov = bias[3 * H_ + n];

    v8f creg = {};                        // cell state: registers for all T steps

    const float* xrow = x + (size_t)m * T_ * D_;

    // Anti-LICM: a zero offset that the empty asm "redefines" every
    // iteration.  Weight-fragment addresses become loop-variant (so 48
    // fragments = 384 VGPRs are NOT hoisted out of the T-loop and spilled
    // to scratch), while indexing off the original kernel-arg pointers
    // keeps global address-space provenance -> global_load_b128, not
    // flat_load (flat would tie up DScnt alongside the LDS h-stream).
    int lv = 0;

#pragma unroll 1
    for (int t = 0; t < T_; ++t) {
        asm volatile("" : "+s"(lv));      // zero-cost; defeats LICM only
        const __bf16* wxTl = wxT + lv;
        const __bf16* whTl = whT + lv;

        v8f acc[4] = {};                  // g, i, f, o pre-activations

        // Prefetch next step's x row (gfx1250 global_prefetch); uniform guard.
        if (t + 1 < T_) __builtin_prefetch(xrow + (size_t)(t + 1) * D_, 0, 1);

        // ---- input projection: K = D = 128 ----
        const float* xp = xrow + (size_t)t * D_;
#pragma unroll
        for (int kb = 0; kb < D_; kb += 32) {
            v16bf a = load_a_f32(xp, kb, lhalf);
#pragma unroll
            for (int g = 0; g < 4; ++g) {
                v16bf bm = load_b(wxTl, D_, g * H_ + n0 + lidx, kb, lhalf);
                acc[g] = __builtin_amdgcn_wmma_f32_16x16x32_bf16(
                    false, a, false, bm, (short)0, acc[g], false, false);
            }
        }

        // ---- recurrent projection from LDS: K = H = 256 ----
        const __bf16* hrow = &hbuf[t & 1][mloc * H_];
#pragma unroll
        for (int kb = 0; kb < H_; kb += 32) {
            v16bf a = load_a_bf16(hrow, kb, lhalf);
#pragma unroll
            for (int g = 0; g < 4; ++g) {
                v16bf bm = load_b(whTl, H_, g * H_ + n0 + lidx, kb, lhalf);
                acc[g] = __builtin_amdgcn_wmma_f32_16x16x32_bf16(
                    false, a, false, bm, (short)0, acc[g], false, false);
            }
        }

        // ---- gates + state update; write h_{t+1} tile into LDS ----
        __bf16* hout = &hbuf[(t + 1) & 1][0];
#pragma unroll
        for (int r = 0; r < 8; ++r) {
            const int mr = r + lhalf * 8;             // C/D fragment row map
            const float g  = tanhf_fast(acc[0][r] + bgv);
            const float ig = sigmoidf_(acc[1][r] + biv);
            const float fg = sigmoidf_(acc[2][r] + bfv);
            const float og = sigmoidf_(acc[3][r] + bov);
            const float cn = g * ig + creg[r] * fg;
            creg[r] = cn;
            hout[mr * H_ + n] = (__bf16)(tanhf_fast(cn) * og);
        }
        __syncthreads();   // one barrier per step (double buffer makes it safe)
    }

    // ---- classifier head: out[m0..m0+15][0..9] from hT in hbuf[0] ----
    if (tid < 16 * C_) {
        const int b   = tid / C_;
        const int cls = tid % C_;
        float s = bp[cls];
#pragma unroll 8
        for (int k = 0; k < H_; ++k)
            s += (float)hbuf[T_ & 1][b * H_ + k] * wph[(size_t)k * C_ + cls];
        out[(size_t)(m0 + b) * C_ + cls] = s;
    }
}

// ---- Prep: transpose+concat gate weights into [4H][K] bf16 --------------
__global__ void prep_wT(const float* __restrict__ w0, const float* __restrict__ w1,
                        const float* __restrict__ w2, const float* __restrict__ w3,
                        __bf16* __restrict__ wT, int K)
{
    int idx = blockIdx.x * blockDim.x + threadIdx.x;   // over 4H*K
    if (idx >= FH_ * K) return;
    int ng = idx / K, k = idx % K;
    int gate = ng / H_, col = ng % H_;
    const float* wsrc = (gate == 0) ? w0 : (gate == 1) ? w1 : (gate == 2) ? w2 : w3;
    wT[idx] = (__bf16)wsrc[(size_t)k * H_ + col];
}

// ---- Prep: concat biases -------------------------------------------------
__global__ void prep_bias(const float* __restrict__ bg, const float* __restrict__ bi,
                          const float* __restrict__ bf, const float* __restrict__ bo,
                          float* __restrict__ bias)
{
    int idx = blockIdx.x * blockDim.x + threadIdx.x;
    if (idx >= FH_) return;
    float v = (idx < H_)     ? bg[idx]
            : (idx < 2 * H_) ? bi[idx - H_]
            : (idx < 3 * H_) ? bf[idx - 2 * H_]
                             : bo[idx - 3 * H_];
    bias[idx] = v;
}

extern "C" void kernel_launch(void* const* d_in, const int* in_sizes, int n_in,
                              void* d_out, int out_size, void* d_ws, size_t ws_size,
                              hipStream_t stream)
{
    const float* x   = (const float*)d_in[0];
    const float* wgx = (const float*)d_in[1];
    const float* wgh = (const float*)d_in[2];
    const float* bg  = (const float*)d_in[3];
    const float* wix = (const float*)d_in[4];
    const float* wih = (const float*)d_in[5];
    const float* bi  = (const float*)d_in[6];
    const float* wfx = (const float*)d_in[7];
    const float* wfh = (const float*)d_in[8];
    const float* bf  = (const float*)d_in[9];
    const float* wox = (const float*)d_in[10];
    const float* woh = (const float*)d_in[11];
    const float* bo  = (const float*)d_in[12];
    const float* wph = (const float*)d_in[13];
    const float* bp  = (const float*)d_in[14];

    // Workspace carve-out (~0.8 MB, all sections 256B-aligned)
    char* ws = (char*)d_ws;
    __bf16* wxT  = (__bf16*)ws;  ws += (size_t)FH_ * D_ * 2;   // [4H][D] bf16
    __bf16* whT  = (__bf16*)ws;  ws += (size_t)FH_ * H_ * 2;   // [4H][H] bf16
    float*  bias = (float*)ws;   ws += (size_t)FH_ * 4;        // [4H] f32

    prep_wT<<<(FH_ * D_ + 255) / 256, 256, 0, stream>>>(wgx, wix, wfx, wox, wxT, D_);
    prep_wT<<<(FH_ * H_ + 255) / 256, 256, 0, stream>>>(wgh, wih, wfh, woh, whT, H_);
    prep_bias<<<(FH_ + 255) / 256, 256, 0, stream>>>(bg, bi, bf, bo, bias);

    // One launch for the whole sequence: 16 independent batch-row blocks,
    // 512 threads (16 waves) each; h lives in LDS, c lives in registers.
    lstm_seq_kernel<<<B_ / 16, 512, 0, stream>>>(x, wxT, whT, bias, wph, bp,
                                                 (float*)d_out);
}
